// GBNModel_86260123173249
// MI455X (gfx1250) — matrix operand order repaired and checked by reading, not compiled, not run on hardware
//
#include <hip/hip_runtime.h>
#include <math.h>

// ---------------- constants ----------------
#define RMINV 2.2e-20f
#define EGAM  0.5772f
// B=128, J=32, V=5000, K1=256, K2=128, K3=64, R=B*J=4096
#define R_TOK 4096
#define VDIM  5000
#define KP1   5024   // V padded to multiple of 32 (K dim of GEMM1)
#define VP1   5008   // V padded to multiple of 16 (N dim of GEMM2)

typedef __bf16 bf16;
typedef __attribute__((ext_vector_type(16))) __bf16        v16bf;
typedef __attribute__((ext_vector_type(8)))  float         v8f;
typedef __attribute__((ext_vector_type(4)))  unsigned int  u32x4;

union FragBF { v16bf v; u32x4 q[2]; };

__device__ __forceinline__ float sigmoidf_(float x) { return 1.f / (1.f + expf(-x)); }
__device__ __forceinline__ float softplusf_(float x) { return (x > 20.f) ? x : log1pf(expf(x)); }

// ---------------- conversion / pad kernels ----------------
__global__ void cvt_ctx_kernel(const float* __restrict__ x, bf16* __restrict__ y) {
    size_t i = (size_t)blockIdx.x * blockDim.x + threadIdx.x;
    size_t total = (size_t)R_TOK * KP1;
    if (i >= total) return;
    size_t r = i / KP1, c = i % KP1;
    y[i] = (c < VDIM) ? (bf16)x[r * VDIM + c] : (bf16)0.f;
}

__global__ void cvt_ws1t_kernel(const float* __restrict__ w, bf16* __restrict__ y) {
    size_t i = (size_t)blockIdx.x * blockDim.x + threadIdx.x;
    size_t total = (size_t)256 * KP1;
    if (i >= total) return;
    size_t n = i / KP1, k = i % KP1;
    y[i] = (k < VDIM) ? (bf16)w[k * 256 + n] : (bf16)0.f;   // Ws1 is [V,K1]; store transposed
}

__global__ void cvt_phi1_kernel(const float* __restrict__ p, bf16* __restrict__ y) {
    size_t i = (size_t)blockIdx.x * blockDim.x + threadIdx.x;
    size_t total = (size_t)VP1 * 256;
    if (i >= total) return;
    size_t v = i / 256, k = i % 256;
    y[i] = (v < VDIM) ? (bf16)p[v * 256 + k] : (bf16)0.f;   // Phi1 is [V,K1], row-major kept
}

// ---------------- GEMM1: s1 = sigmoid(context @ Ws1), bf16 WMMA ----------------
// A: ctx_bf [4096][5024], B(as Bt rows=N): ws1t_bf [256][5024], out s1 f32 [4096][256]
__global__ __launch_bounds__(32) void gemm1_wmma_kernel(
    const bf16* __restrict__ A, const bf16* __restrict__ Bt, float* __restrict__ s1) {
    const int lane = threadIdx.x & 31;
    const int half = lane >> 4;
    const int lm   = lane & 15;
    const int mt = blockIdx.x;            // 0..255  (M tile)
    const int ntB = blockIdx.y * 4;       // 4 N tiles per wave

    const bf16* arow  = A  + (size_t)(mt * 16 + lm) * KP1;
    const bf16* brow0 = Bt + (size_t)((ntB + 0) * 16 + lm) * KP1 + half * 16;
    const bf16* brow1 = Bt + (size_t)((ntB + 1) * 16 + lm) * KP1 + half * 16;
    const bf16* brow2 = Bt + (size_t)((ntB + 2) * 16 + lm) * KP1 + half * 16;
    const bf16* brow3 = Bt + (size_t)((ntB + 3) * 16 + lm) * KP1 + half * 16;

    v8f acc0 = {}, acc1 = {}, acc2 = {}, acc3 = {};
    for (int k0 = 0; k0 < KP1; k0 += 32) {
        FragBF a, b0, b1, b2, b3;
        a.q[0]  = *(const u32x4*)(arow + k0 + half * 16);
        a.q[1]  = *(const u32x4*)(arow + k0 + 8 + half * 16);
        b0.q[0] = *(const u32x4*)(brow0 + k0);  b0.q[1] = *(const u32x4*)(brow0 + k0 + 8);
        b1.q[0] = *(const u32x4*)(brow1 + k0);  b1.q[1] = *(const u32x4*)(brow1 + k0 + 8);
        b2.q[0] = *(const u32x4*)(brow2 + k0);  b2.q[1] = *(const u32x4*)(brow2 + k0 + 8);
        b3.q[0] = *(const u32x4*)(brow3 + k0);  b3.q[1] = *(const u32x4*)(brow3 + k0 + 8);
        acc0 = __builtin_amdgcn_wmma_f32_16x16x32_bf16(false, a.v, false, b0.v, (short)0, acc0, false, false);
        acc1 = __builtin_amdgcn_wmma_f32_16x16x32_bf16(false, a.v, false, b1.v, (short)0, acc1, false, false);
        acc2 = __builtin_amdgcn_wmma_f32_16x16x32_bf16(false, a.v, false, b2.v, (short)0, acc2, false, false);
        acc3 = __builtin_amdgcn_wmma_f32_16x16x32_bf16(false, a.v, false, b3.v, (short)0, acc3, false, false);
    }
#pragma unroll
    for (int r = 0; r < 8; ++r) {
        size_t mm = (size_t)mt * 16 + r + half * 8;
        s1[mm * 256 + (ntB + 0) * 16 + lm] = sigmoidf_(acc0[r]);
        s1[mm * 256 + (ntB + 1) * 16 + lm] = sigmoidf_(acc1[r]);
        s1[mm * 256 + (ntB + 2) * 16 + lm] = sigmoidf_(acc2[r]);
        s1[mm * 256 + (ntB + 3) * 16 + lm] = sigmoidf_(acc3[r]);
    }
}

// ---------------- GEMM2 fused: Px = t1 @ Phi1^T; L1 = ctx*log(max(Px)) - Px ----------------
__global__ __launch_bounds__(32) void gemm2_l1_kernel(
    const bf16* __restrict__ T1,   // [4096][256] bf16
    const bf16* __restrict__ P1,   // [5008][256] bf16 (Phi1 rows, zero padded)
    const float* __restrict__ ctx, // [4096][5000] f32
    double* __restrict__ l1acc) {
    const int lane = threadIdx.x & 31;
    const int half = lane >> 4;
    const int lm   = lane & 15;
    const int mt = blockIdx.x;   // 0..255
    const int nt = blockIdx.y;   // 0..312

    const bf16* arow = T1 + (size_t)(mt * 16 + lm) * 256 + half * 16;
    const bf16* brow = P1 + (size_t)(nt * 16 + lm) * 256 + half * 16;

    v8f acc = {};
#pragma unroll
    for (int k0 = 0; k0 < 256; k0 += 32) {
        FragBF a, b;
        a.q[0] = *(const u32x4*)(arow + k0);
        a.q[1] = *(const u32x4*)(arow + k0 + 8);
        b.q[0] = *(const u32x4*)(brow + k0);
        b.q[1] = *(const u32x4*)(brow + k0 + 8);
        acc = __builtin_amdgcn_wmma_f32_16x16x32_bf16(false, a.v, false, b.v, (short)0, acc, false, false);
    }
    float lsum = 0.f;
    int nn = nt * 16 + lm;
    if (nn < VDIM) {
#pragma unroll
        for (int r = 0; r < 8; ++r) {
            size_t mm = (size_t)mt * 16 + r + half * 8;
            float px = acc[r];
            float c  = ctx[mm * VDIM + nn];
            lsum += c * logf(fmaxf(px, RMINV)) - px;
        }
    }
    for (int off = 16; off > 0; off >>= 1) lsum += __shfl_down(lsum, off, 32);
    if (lane == 0) atomicAdd(l1acc, (double)lsum);
}

// ---------------- s2 = sigmoid(s1@Ws2); s3 = sigmoid(s2@Ws3) ----------------
__global__ void s2s3_kernel(const float* __restrict__ s1, const float* __restrict__ Ws2,
                            const float* __restrict__ Ws3, float* __restrict__ s2,
                            float* __restrict__ s3) {
    __shared__ float r1[256];
    __shared__ float r2[128];
    int row = blockIdx.x, tid = threadIdx.x;     // blockDim = 128
    r1[tid]       = s1[(size_t)row * 256 + tid];
    r1[tid + 128] = s1[(size_t)row * 256 + tid + 128];
    __syncthreads();
    float a = 0.f;
    for (int q = 0; q < 256; ++q) a += r1[q] * Ws2[q * 128 + tid];
    float v = sigmoidf_(a);
    r2[tid] = v;
    s2[(size_t)row * 128 + tid] = v;
    __syncthreads();
    if (tid < 64) {
        float a3 = 0.f;
        for (int q = 0; q < 128; ++q) a3 += r2[q] * Ws3[q * 64 + tid];
        s3[(size_t)row * 64 + tid] = sigmoidf_(a3);
    }
}

// ---------------- encoder theta per level + ht output ----------------
__global__ void theta_kernel(const float* __restrict__ s, const float* __restrict__ Wk,
                             const float* __restrict__ bk, const float* __restrict__ Wl,
                             const float* __restrict__ bl, const float* __restrict__ eps,
                             float* __restrict__ kout, float* __restrict__ lam,
                             float* __restrict__ theta, bf16* __restrict__ theta_bf,
                             float* __restrict__ ht, int K) {
    __shared__ float srow[256];
    __shared__ float red[256];
    int row = blockIdx.x, tid = threadIdx.x;     // blockDim = K
    float sv = s[(size_t)row * K + tid];
    srow[tid] = sv;
    red[tid]  = sv * Wk[tid];
    __syncthreads();
    for (int st = K >> 1; st > 0; st >>= 1) { if (tid < st) red[tid] += red[tid + st]; __syncthreads(); }
    float kval = fmaxf(softplusf_(red[0] + bk[0]), RMINV);

    float lacc = bl[tid];
    for (int q = 0; q < K; ++q) lacc += srow[q] * Wl[q * K + tid];
    float l = softplusf_(lacc);
    float e = eps[(size_t)row * K + tid];
    float w = powf(-logf(fmaxf(1.f - e, RMINV)), 1.f / kval);
    float th = l * w;
    lam[(size_t)row * K + tid]   = l;
    theta[(size_t)row * K + tid] = th;
    if (theta_bf) theta_bf[(size_t)row * K + tid] = (bf16)th;
    if (tid == 0) kout[row] = kval;

    __syncthreads();
    red[tid] = th;
    __syncthreads();
    for (int st = K >> 1; st > 0; st >>= 1) { if (tid < st) red[tid] = fmaxf(red[tid], red[tid + st]); __syncthreads(); }
    float mx = fmaxf(red[0], RMINV);
    int b = row >> 5, j = row & 31;              // row = b*32 + j
    ht[((size_t)(j * 128 + b)) * K + tid] = th / mx;   // j-major [J*B, K]
}

// ---------------- alpha recurrence ----------------
__global__ void alpha_kernel(const float* __restrict__ Phi, const float* __restrict__ tn, int Kn,
                             const float* __restrict__ Pi, const float* __restrict__ tc,
                             float* __restrict__ aout, int K, int onesJ0) {
    __shared__ float shn[256];
    __shared__ float shp[256];
    int row = blockIdx.x, tid = threadIdx.x;     // blockDim = K
    int j = row & 31;
    if (Phi && tid < Kn) shn[tid] = tn[(size_t)row * Kn + tid];
    if (j > 0) shp[tid] = tc[(size_t)(row - 1) * K + tid];
    __syncthreads();
    float acc = 0.f;
    if (Phi) { for (int q = 0; q < Kn; ++q) acc += Phi[tid * Kn + q] * shn[q]; }
    if (j > 0) { for (int q = 0; q < K; ++q) acc += Pi[tid * K + q] * shp[q]; }
    if (onesJ0 && j == 0) acc = 1.f;
    aout[(size_t)row * K + tid] = acc;
}

// ---------------- KL(Gamma || Weibull) sum ----------------
__global__ void kl_kernel(const float* __restrict__ alpha, const float* __restrict__ kk,
                          const float* __restrict__ lam, int K, int total,
                          double* __restrict__ acc) {
    int idx = blockIdx.x * blockDim.x + threadIdx.x;
    float sum = 0.f;
    for (int i = idx; i < total; i += gridDim.x * blockDim.x) {
        int row = i / K;
        float k = kk[row];
        float a = alpha[i];
        float l = lam[i];
        float invk = 1.f / k;
        float p1 = EGAM * (1.f - invk) + logf(fmaxf(l * invk, RMINV)) + 1.f;
        float p2 = -lgammaf(a) + (a - 1.f) * (logf(fmaxf(l, RMINV)) - EGAM * invk);
        sum += p1 + p2 - l * expf(lgammaf(1.f + invk));
    }
    for (int off = 16; off > 0; off >>= 1) sum += __shfl_down(sum, off, 32);
    if ((threadIdx.x & 31) == 0) atomicAdd(acc, (double)sum);
}

// ---------------- scalar plumbing ----------------
__global__ void zero_acc_kernel(double* acc) { if (threadIdx.x < 4) acc[threadIdx.x] = 0.0; }

__global__ void finalize_kernel(const double* __restrict__ acc, float* __restrict__ out) {
    if (threadIdx.x == 0)
        out[0] = (float)((acc[0] + 0.1 * acc[1] + 0.01 * acc[2] + 0.001 * acc[3]) / 128.0);
}

// ---------------- launch ----------------
extern "C" void kernel_launch(void* const* d_in, const int* in_sizes, int n_in,
                              void* d_out, int out_size, void* d_ws, size_t ws_size,
                              hipStream_t stream) {
    (void)in_sizes; (void)n_in; (void)out_size; (void)ws_size;
    const float* ctx  = (const float*)d_in[0];
    const float* eps1 = (const float*)d_in[1];
    const float* eps2 = (const float*)d_in[2];
    const float* eps3 = (const float*)d_in[3];
    const float* Phi1 = (const float*)d_in[4];
    const float* Phi2 = (const float*)d_in[5];
    const float* Phi3 = (const float*)d_in[6];
    const float* Pi1  = (const float*)d_in[7];
    const float* Pi2  = (const float*)d_in[8];
    const float* Pi3  = (const float*)d_in[9];
    const float* Ws1  = (const float*)d_in[10];
    const float* Ws2  = (const float*)d_in[11];
    const float* Ws3  = (const float*)d_in[12];
    const float* Wk1  = (const float*)d_in[13];
    const float* bk1  = (const float*)d_in[14];
    const float* Wl1  = (const float*)d_in[15];
    const float* bl1  = (const float*)d_in[16];
    const float* Wk2  = (const float*)d_in[17];
    const float* bk2  = (const float*)d_in[18];
    const float* Wl2  = (const float*)d_in[19];
    const float* bl2  = (const float*)d_in[20];
    const float* Wk3  = (const float*)d_in[21];
    const float* bk3  = (const float*)d_in[22];
    const float* Wl3  = (const float*)d_in[23];
    const float* bl3  = (const float*)d_in[24];
    float* out = (float*)d_out;

    char* p = (char*)d_ws;
    auto carve = [&](size_t bytes) -> char* {
        char* r = p; p += (bytes + 255) & ~(size_t)255; return r;
    };
    double* acc   = (double*)carve(4 * sizeof(double));
    bf16* ctx_bf  = (bf16*)carve((size_t)R_TOK * KP1 * 2);
    bf16* ws1t_bf = (bf16*)carve((size_t)256 * KP1 * 2);
    bf16* phi1_bf = (bf16*)carve((size_t)VP1 * 256 * 2);
    bf16* t1_bf   = (bf16*)carve((size_t)R_TOK * 256 * 2);
    float* s1 = (float*)carve((size_t)R_TOK * 256 * 4);
    float* s2 = (float*)carve((size_t)R_TOK * 128 * 4);
    float* s3 = (float*)carve((size_t)R_TOK * 64 * 4);
    float* k1 = (float*)carve((size_t)R_TOK * 4);
    float* k2 = (float*)carve((size_t)R_TOK * 4);
    float* k3 = (float*)carve((size_t)R_TOK * 4);
    float* l1 = (float*)carve((size_t)R_TOK * 256 * 4);
    float* l2 = (float*)carve((size_t)R_TOK * 128 * 4);
    float* l3 = (float*)carve((size_t)R_TOK * 64 * 4);
    float* t1 = (float*)carve((size_t)R_TOK * 256 * 4);
    float* t2 = (float*)carve((size_t)R_TOK * 128 * 4);
    float* t3 = (float*)carve((size_t)R_TOK * 64 * 4);
    float* a1 = (float*)carve((size_t)R_TOK * 256 * 4);
    float* a2 = (float*)carve((size_t)R_TOK * 128 * 4);
    float* a3 = (float*)carve((size_t)R_TOK * 64 * 4);

    float* ht1 = out + 1;
    float* ht2 = ht1 + (size_t)R_TOK * 256;
    float* ht3 = ht2 + (size_t)R_TOK * 128;

    zero_acc_kernel<<<1, 32, 0, stream>>>(acc);

    cvt_ctx_kernel <<<((size_t)R_TOK * KP1 + 255) / 256, 256, 0, stream>>>(ctx, ctx_bf);
    cvt_ws1t_kernel<<<((size_t)256 * KP1 + 255) / 256, 256, 0, stream>>>(Ws1, ws1t_bf);
    cvt_phi1_kernel<<<((size_t)VP1 * 256 + 255) / 256, 256, 0, stream>>>(Phi1, phi1_bf);

    gemm1_wmma_kernel<<<dim3(256, 4), 32, 0, stream>>>(ctx_bf, ws1t_bf, s1);

    s2s3_kernel<<<R_TOK, 128, 0, stream>>>(s1, Ws2, Ws3, s2, s3);

    theta_kernel<<<R_TOK, 256, 0, stream>>>(s1, Wk1, bk1, Wl1, bl1, eps1, k1, l1, t1, t1_bf, ht1, 256);
    theta_kernel<<<R_TOK, 128, 0, stream>>>(s2, Wk2, bk2, Wl2, bl2, eps2, k2, l2, t2, nullptr, ht2, 128);
    theta_kernel<<<R_TOK,  64, 0, stream>>>(s3, Wk3, bk3, Wl3, bl3, eps3, k3, l3, t3, nullptr, ht3, 64);

    alpha_kernel<<<R_TOK, 256, 0, stream>>>(Phi2, t2, 128, Pi1, t1, a1, 256, 0);
    alpha_kernel<<<R_TOK, 128, 0, stream>>>(Phi3, t3,  64, Pi2, t2, a2, 128, 0);
    alpha_kernel<<<R_TOK,  64, 0, stream>>>(nullptr, nullptr, 0, Pi3, t3, a3, 64, 1);

    kl_kernel<<<1024, 256, 0, stream>>>(a1, k1, l1, 256, R_TOK * 256, acc + 1);
    kl_kernel<<<1024, 256, 0, stream>>>(a2, k2, l2, 128, R_TOK * 128, acc + 2);
    kl_kernel<<<1024, 256, 0, stream>>>(a3, k3, l3,  64, R_TOK * 64,  acc + 3);

    gemm2_l1_kernel<<<dim3(256, 313), 32, 0, stream>>>(t1_bf, phi1_bf, ctx, acc);

    finalize_kernel<<<1, 32, 0, stream>>>(acc, out);
}
